// GATFraudDetector_20323785244836
// MI455X (gfx1250) — matrix operand order repaired and checked by reading, not compiled, not run on hardware
//
#include <hip/hip_runtime.h>

// ---------------------------------------------------------------------------
// GAT fraud detector for MI455X (gfx1250), wave32, WMMA bf16 GEMMs.
// ---------------------------------------------------------------------------

typedef unsigned int   u32;
typedef unsigned short u16;

typedef __attribute__((ext_vector_type(16))) __bf16 v16bf;
typedef __attribute__((ext_vector_type(8)))  float  v8f;

union FragB { v16bf v; uint4 q[2]; };

#define NEG_SLOPE 0.2f
#define BN_INV 0.99999500003749984f   // 1/sqrt(1+1e-5)

__device__ __forceinline__ u16 f32_to_bf16(float f) {
  u32 u = __float_as_uint(f);
  u32 r = u + 0x7fffu + ((u >> 16) & 1u);   // round-to-nearest-even
  return (u16)(r >> 16);
}

// monotone order-preserving float -> uint key (for atomic max)
__device__ __forceinline__ u32 fkey(float f) {
  u32 u = __float_as_uint(f);
  return (u & 0x80000000u) ? ~u : (u | 0x80000000u);
}
__device__ __forceinline__ float funkey(u32 k) {
  u32 u = (k & 0x80000000u) ? (k & 0x7fffffffu) : ~k;
  return __uint_as_float(u);
}

__device__ __forceinline__ void atomAdd(float* p, float v) {
  __hip_atomic_fetch_add(p, v, __ATOMIC_RELAXED, __HIP_MEMORY_SCOPE_AGENT);
}

// ---------------------------------------------------------------------------
__global__ void k_f32_to_bf16(const float* __restrict__ src, u16* __restrict__ dst,
                              long long n) {
  long long i = (long long)blockIdx.x * blockDim.x + threadIdx.x;
  if (i < n) dst[i] = f32_to_bf16(src[i]);
}

__global__ void k_zero(float* __restrict__ p, long long n) {
  long long i = (long long)blockIdx.x * blockDim.x + threadIdx.x;
  if (i < n) p[i] = 0.f;
}

// ---------------------------------------------------------------------------
// NT GEMM: D[M,256] = A[M,K] * B[256,K]^T, A/B bf16 row-major, D f32.
// One wave computes a 16x64 strip: one A fragment reused across 4 B subtiles
// (4x v_wmma_f32_16x16x32_bf16 per K-step). All B/store addresses are
// immediate offsets from one base (32-bit element offsets, O=256 is constexpr).
//
// A frag (16x32, lane m=lane&15, half=lane>>4): K chunk  [16h, 16h+16)
// B frag (32x16, lane n=lane&15, half=lane>>4): K chunk  [16h, 16h+16)
// C/D layout: VGPR j -> row = tm + j + 8*half, col = tn + (lane&15)
// ---------------------------------------------------------------------------
template <int K>
__global__ void k_gemm_bf16_nt(const u16* __restrict__ A, const u16* __restrict__ B,
                               float* __restrict__ D, int M) {
  constexpr int O = 256;          // output width (H*C)
  constexpr int NG = O / 64;      // 4 N-groups of 64 columns
  int wave = (int)((blockIdx.x * blockDim.x + threadIdx.x) >> 5);
  int lane = threadIdx.x & 31;
  int tilesM = M >> 4;
  if (wave >= tilesM * NG) return;   // wave-uniform: EXEC stays all-ones
  int tm = (wave / NG) << 4;
  int tn = (wave % NG) << 6;
  int half = lane >> 4;
  int r    = lane & 15;
  u32 aoff = (u32)(tm + r) * K + 16 * half;
  u32 boff = (u32)(tn + r) * K + 16 * half;
  v8f c0 = {}, c1 = {}, c2 = {}, c3 = {};
#pragma unroll
  for (int kk = 0; kk < K; kk += 32) {
    FragB a, b0, b1, b2, b3;
    a.q[0]  = *(const uint4*)(A + aoff + kk);
    a.q[1]  = *(const uint4*)(A + aoff + kk + 8);
    b0.q[0] = *(const uint4*)(B + boff + kk);
    b0.q[1] = *(const uint4*)(B + boff + kk + 8);
    b1.q[0] = *(const uint4*)(B + boff + 16 * K + kk);
    b1.q[1] = *(const uint4*)(B + boff + 16 * K + kk + 8);
    b2.q[0] = *(const uint4*)(B + boff + 32 * K + kk);
    b2.q[1] = *(const uint4*)(B + boff + 32 * K + kk + 8);
    b3.q[0] = *(const uint4*)(B + boff + 48 * K + kk);
    b3.q[1] = *(const uint4*)(B + boff + 48 * K + kk + 8);
    c0 = __builtin_amdgcn_wmma_f32_16x16x32_bf16(false, a.v, false, b0.v,
                                                 (short)0, c0, false, false);
    c1 = __builtin_amdgcn_wmma_f32_16x16x32_bf16(false, a.v, false, b1.v,
                                                 (short)0, c1, false, false);
    c2 = __builtin_amdgcn_wmma_f32_16x16x32_bf16(false, a.v, false, b2.v,
                                                 (short)0, c2, false, false);
    c3 = __builtin_amdgcn_wmma_f32_16x16x32_bf16(false, a.v, false, b3.v,
                                                 (short)0, c3, false, false);
  }
  u32 doff = (u32)(tm + 8 * half) * O + tn + r;
#pragma unroll
  for (int j = 0; j < 8; ++j) {
    D[doff + j * O +  0] = c0[j];
    D[doff + j * O + 16] = c1[j];
    D[doff + j * O + 32] = c2[j];
    D[doff + j * O + 48] = c3[j];
  }
}

// ---------------------------------------------------------------------------
// Per-node attention partials: ssrc[n,h] = dot(feat[n,h*64:..], a_src[h]), same dst.
// Block = 128 threads = 4 waves = 4 heads of one node. Also init mcode/denom.
// ---------------------------------------------------------------------------
__global__ void k_node_attn(const float* __restrict__ feat,
                            const float* __restrict__ asrc, const float* __restrict__ adst,
                            float* __restrict__ ss, float* __restrict__ sd,
                            u32* __restrict__ mc, float* __restrict__ den, int Nn) {
  int n = blockIdx.x;
  if (n >= Nn) return;
  int t = threadIdx.x;
  int h = t >> 5;       // head (wave per head)
  int l = t & 31;
  const float2 f = *(const float2*)(feat + (size_t)n * 256 + h * 64 + l * 2);
  const float2 s = *(const float2*)(asrc + h * 64 + l * 2);
  const float2 d = *(const float2*)(adst + h * 64 + l * 2);
  float vs = f.x * s.x + f.y * s.y;
  float vd = f.x * d.x + f.y * d.y;
  for (int off = 16; off; off >>= 1) { vs += __shfl_xor(vs, off); vd += __shfl_xor(vd, off); }
  if (l == 0) {
    ss[(size_t)n * 4 + h] = vs;
    sd[(size_t)n * 4 + h] = vd;
    mc[(size_t)n * 4 + h] = 0u;      // below fkey() of any finite float
    den[(size_t)n * 4 + h] = 0.f;
  }
}

// ---------------------------------------------------------------------------
// Edge pass 1: segment max of leaky_relu(ssrc[src]+sdst[dst]) per (dst, head).
// Self-loops are edges [E, E+N).
// ---------------------------------------------------------------------------
__global__ void k_edge_max(const int* __restrict__ src, const int* __restrict__ dst,
                           const float* __restrict__ ss, const float* __restrict__ sd,
                           u32* __restrict__ mc, int E, int Nn) {
  int e = blockIdx.x * blockDim.x + threadIdx.x;
  if (e >= E + Nn) return;
  int s = (e < E) ? src[e] : e - E;
  int d = (e < E) ? dst[e] : e - E;
  float4 a = *(const float4*)(ss + (size_t)s * 4);
  float4 b = *(const float4*)(sd + (size_t)d * 4);
  u32* m = mc + (size_t)d * 4;
  float v;
  v = a.x + b.x; v = v > 0.f ? v : NEG_SLOPE * v; atomicMax(m + 0, fkey(v));
  v = a.y + b.y; v = v > 0.f ? v : NEG_SLOPE * v; atomicMax(m + 1, fkey(v));
  v = a.z + b.z; v = v > 0.f ? v : NEG_SLOPE * v; atomicMax(m + 2, fkey(v));
  v = a.w + b.w; v = v > 0.f ? v : NEG_SLOPE * v; atomicMax(m + 3, fkey(v));
}

// Edge pass 2: ex = exp(e - m[dst]); store ex, accumulate denom.
__global__ void k_edge_expsum(const int* __restrict__ src, const int* __restrict__ dst,
                              const float* __restrict__ ss, const float* __restrict__ sd,
                              const u32* __restrict__ mc, float* __restrict__ den,
                              float* __restrict__ ex, int E, int Nn) {
  int e = blockIdx.x * blockDim.x + threadIdx.x;
  if (e >= E + Nn) return;
  int s = (e < E) ? src[e] : e - E;
  int d = (e < E) ? dst[e] : e - E;
  float4 a = *(const float4*)(ss + (size_t)s * 4);
  float4 b = *(const float4*)(sd + (size_t)d * 4);
  const u32* m = mc + (size_t)d * 4;
  float4 o; float v;
  v = a.x + b.x; v = v > 0.f ? v : NEG_SLOPE * v; o.x = __expf(v - funkey(m[0]));
  v = a.y + b.y; v = v > 0.f ? v : NEG_SLOPE * v; o.y = __expf(v - funkey(m[1]));
  v = a.z + b.z; v = v > 0.f ? v : NEG_SLOPE * v; o.z = __expf(v - funkey(m[2]));
  v = a.w + b.w; v = v > 0.f ? v : NEG_SLOPE * v; o.w = __expf(v - funkey(m[3]));
  *(float4*)(ex + (size_t)e * 4) = o;
  float* dn = den + (size_t)d * 4;
  atomAdd(dn + 0, o.x); atomAdd(dn + 1, o.y); atomAdd(dn + 2, o.z); atomAdd(dn + 3, o.w);
}

// Edge pass 3: agg[dst] += feat[src] * alpha. One wave per edge, 8 ch/lane.
__global__ void k_edge_agg(const int* __restrict__ src, const int* __restrict__ dst,
                           const float* __restrict__ feat, const float* __restrict__ ex,
                           const float* __restrict__ den, float* __restrict__ agg,
                           int E, int Nn) {
  long long gid = (long long)blockIdx.x * blockDim.x + threadIdx.x;
  int gw = (int)(gid >> 5);
  int lane = threadIdx.x & 31;
  if (gw >= E + Nn) return;
  int s = (gw < E) ? src[gw] : gw - E;
  int d = (gw < E) ? dst[gw] : gw - E;
  int f0 = lane * 8;
  int h = f0 >> 6;
  float alpha = ex[(size_t)gw * 4 + h] / (den[(size_t)d * 4 + h] + 1e-16f);
  const float4* fs = (const float4*)(feat + (size_t)s * 256 + f0);
  float4 v0 = fs[0], v1 = fs[1];
  float* ad = agg + (size_t)d * 256 + f0;
  atomAdd(ad + 0, v0.x * alpha); atomAdd(ad + 1, v0.y * alpha);
  atomAdd(ad + 2, v0.z * alpha); atomAdd(ad + 3, v0.w * alpha);
  atomAdd(ad + 4, v1.x * alpha); atomAdd(ad + 5, v1.y * alpha);
  atomAdd(ad + 6, v1.z * alpha); atomAdd(ad + 7, v1.w * alpha);
}

// Bias + eval-mode BN + ReLU; emit bf16 (next GEMM) and/or f32.
__global__ void k_finalize(const float* __restrict__ agg, const float* __restrict__ bias,
                           const float* __restrict__ g, const float* __restrict__ bb,
                           u16* __restrict__ out_bf, float* __restrict__ out_f32,
                           long long total) {
  long long i = (long long)blockIdx.x * blockDim.x + threadIdx.x;
  if (i >= total) return;
  int f = (int)(i & 255);
  float v = agg[i] + bias[f];
  v = g[f] * v * BN_INV + bb[f];
  v = fmaxf(v, 0.f);
  if (out_bf)  out_bf[i]  = f32_to_bf16(v);
  if (out_f32) out_f32[i] = v;
}

// ---------------------------------------------------------------------------
// Layer 3 (heads=1, C=1): feat3[n] = dot(x2[n], W3). Wave per node.
// ---------------------------------------------------------------------------
__global__ void k_node3(const float* __restrict__ x2, const float* __restrict__ W3,
                        const float* __restrict__ asrc, const float* __restrict__ adst,
                        float* __restrict__ f3, float* __restrict__ ss3, float* __restrict__ sd3,
                        u32* __restrict__ mc3, float* __restrict__ den3,
                        float* __restrict__ agg3, int Nn) {
  long long gid = (long long)blockIdx.x * blockDim.x + threadIdx.x;
  int n = (int)(gid >> 5);
  int lane = threadIdx.x & 31;
  if (n >= Nn) return;
  const float* row = x2 + (size_t)n * 256 + lane * 8;
  const float* wr  = W3 + lane * 8;
  float s = 0.f;
#pragma unroll
  for (int i = 0; i < 8; ++i) s += row[i] * wr[i];
  for (int off = 16; off; off >>= 1) s += __shfl_xor(s, off);
  if (lane == 0) {
    f3[n] = s;
    ss3[n] = s * asrc[0];
    sd3[n] = s * adst[0];
    mc3[n] = 0u;
    den3[n] = 0.f;
    agg3[n] = 0.f;
  }
}

__global__ void k_edge_max1(const int* __restrict__ src, const int* __restrict__ dst,
                            const float* __restrict__ ss, const float* __restrict__ sd,
                            u32* __restrict__ mc, int E, int Nn) {
  int e = blockIdx.x * blockDim.x + threadIdx.x;
  if (e >= E + Nn) return;
  int s = (e < E) ? src[e] : e - E;
  int d = (e < E) ? dst[e] : e - E;
  float v = ss[s] + sd[d];
  v = v > 0.f ? v : NEG_SLOPE * v;
  atomicMax(mc + d, fkey(v));
}

__global__ void k_edge_expsum1(const int* __restrict__ src, const int* __restrict__ dst,
                               const float* __restrict__ ss, const float* __restrict__ sd,
                               const u32* __restrict__ mc, float* __restrict__ den,
                               float* __restrict__ ex, int E, int Nn) {
  int e = blockIdx.x * blockDim.x + threadIdx.x;
  if (e >= E + Nn) return;
  int s = (e < E) ? src[e] : e - E;
  int d = (e < E) ? dst[e] : e - E;
  float v = ss[s] + sd[d];
  v = v > 0.f ? v : NEG_SLOPE * v;
  float o = __expf(v - funkey(mc[d]));
  ex[e] = o;
  atomAdd(den + d, o);
}

__global__ void k_edge_agg1(const int* __restrict__ src, const int* __restrict__ dst,
                            const float* __restrict__ f3, const float* __restrict__ ex,
                            const float* __restrict__ den, float* __restrict__ agg,
                            int E, int Nn) {
  int e = blockIdx.x * blockDim.x + threadIdx.x;
  if (e >= E + Nn) return;
  int s = (e < E) ? src[e] : e - E;
  int d = (e < E) ? dst[e] : e - E;
  float alpha = ex[e] / (den[d] + 1e-16f);
  atomAdd(agg + d, f3[s] * alpha);
}

__global__ void k_reduce(const float* __restrict__ v, float* __restrict__ red, int n) {
  float s = 0.f;
  for (int i = blockIdx.x * blockDim.x + threadIdx.x; i < n; i += gridDim.x * blockDim.x)
    s += v[i];
  for (int off = 16; off; off >>= 1) s += __shfl_xor(s, off);
  __shared__ float sh[8];
  int t = threadIdx.x;
  if ((t & 31) == 0) sh[t >> 5] = s;
  __syncthreads();
  if (t == 0) {
    float acc = 0.f;
    for (int i = 0; i < 8; ++i) acc += sh[i];
    atomAdd(red, acc);
  }
}

// mean -> Linear(1,64) -> ReLU -> Linear(64,1) -> Sigmoid
__global__ void k_classifier(const float* __restrict__ red, const float* __restrict__ b3,
                             const float* __restrict__ cW1, const float* __restrict__ cb1,
                             const float* __restrict__ cW2, const float* __restrict__ cb2,
                             float* __restrict__ out, int Nn) {
  __shared__ float sh[2];
  int t = threadIdx.x;   // 64 threads = 2 waves
  float mean = red[0] / (float)Nn + b3[0];
  float p = fmaxf(mean * cW1[t] + cb1[t], 0.f);
  float v = p * cW2[t];
  for (int off = 16; off; off >>= 1) v += __shfl_xor(v, off);
  if ((t & 31) == 0) sh[t >> 5] = v;
  __syncthreads();
  if (t == 0) {
    float z = sh[0] + sh[1] + cb2[0];
    out[0] = 1.f / (1.f + __expf(-z));
  }
}

// ---------------------------------------------------------------------------
extern "C" void kernel_launch(void* const* d_in, const int* in_sizes, int n_in,
                              void* d_out, int out_size, void* d_ws, size_t ws_size,
                              hipStream_t stream) {
  const float* x     = (const float*)d_in[0];
  const int*   ei    = (const int*)d_in[1];
  const float* W1    = (const float*)d_in[2];
  const float* asrc1 = (const float*)d_in[3];
  const float* adst1 = (const float*)d_in[4];
  const float* b1    = (const float*)d_in[5];
  const float* W2    = (const float*)d_in[6];
  const float* asrc2 = (const float*)d_in[7];
  const float* adst2 = (const float*)d_in[8];
  const float* b2    = (const float*)d_in[9];
  const float* W3    = (const float*)d_in[10];
  const float* asrc3 = (const float*)d_in[11];
  const float* adst3 = (const float*)d_in[12];
  const float* b3    = (const float*)d_in[13];
  const float* bn1g  = (const float*)d_in[14];
  const float* bn1b  = (const float*)d_in[15];
  const float* bn2g  = (const float*)d_in[16];
  const float* bn2b  = (const float*)d_in[17];
  const float* cW1   = (const float*)d_in[18];
  const float* cb1   = (const float*)d_in[19];
  const float* cW2   = (const float*)d_in[20];
  const float* cb2   = (const float*)d_in[21];

  const int N = in_sizes[0] / 128;
  const int E = in_sizes[1] / 2;
  const int* esrc = ei;
  const int* edst = ei + E;

  // ---- workspace carve (256B aligned regions) ----
  char* base = (char*)d_ws;
  size_t off = 0;
  auto carve = [&](size_t bytes) -> void* {
    void* p = base + off;
    off += (bytes + 255) & ~(size_t)255;
    return p;
  };
  u16*   BF   = (u16*)  carve((size_t)N * 256 * 2);       // bf16 GEMM input (reused L1->L2)
  u16*   WB   = (u16*)  carve((size_t)256 * 256 * 2);     // bf16 weights (reused)
  float* FEAT = (float*)carve((size_t)N * 256 * 4);       // h = x@W.T ; later x2 (f32)
  float* AGG  = (float*)carve((size_t)N * 256 * 4);       // aggregated output
  float* SS   = (float*)carve((size_t)N * 4 * 4);
  float* SD   = (float*)carve((size_t)N * 4 * 4);
  u32*   MC   = (u32*)  carve((size_t)N * 4 * 4);
  float* DEN  = (float*)carve((size_t)N * 4 * 4);
  float* EX   = (float*)carve((size_t)(E + N) * 4 * 4);
  float* F3   = (float*)carve((size_t)N * 4);
  float* SS3  = (float*)carve((size_t)N * 4);
  float* SD3  = (float*)carve((size_t)N * 4);
  u32*   MC3  = (u32*)  carve((size_t)N * 4);
  float* DEN3 = (float*)carve((size_t)N * 4);
  float* AGG3 = (float*)carve((size_t)N * 4);
  float* EX3  = (float*)carve((size_t)(E + N) * 4);
  float* RED  = (float*)carve(256 * 4);

  const long long nodeFeat = (long long)N * 256;
  const int nfBlocks = (int)((nodeFeat + 255) / 256);
  const int eb  = (E + N + 255) / 256;                           // thread-per-edge
  const int ab  = (int)(((long long)(E + N) * 32 + 255) / 256);  // wave-per-edge
  const int gemmWaves  = (N / 16) * 4;                           // 16x64 strip per wave
  const int gemmBlocks = (gemmWaves * 32 + 255) / 256;

  auto run_layer = [&](int K, const float* as, const float* ad, const float* bias,
                       const float* g, const float* bb, u16* out_bf, float* out_f32) {
    if (K == 128)
      k_gemm_bf16_nt<128><<<gemmBlocks, 256, 0, stream>>>(BF, WB, FEAT, N);
    else
      k_gemm_bf16_nt<256><<<gemmBlocks, 256, 0, stream>>>(BF, WB, FEAT, N);
    k_node_attn<<<N, 128, 0, stream>>>(FEAT, as, ad, SS, SD, MC, DEN, N);
    k_zero<<<nfBlocks, 256, 0, stream>>>(AGG, nodeFeat);
    k_edge_max   <<<eb, 256, 0, stream>>>(esrc, edst, SS, SD, MC, E, N);
    k_edge_expsum<<<eb, 256, 0, stream>>>(esrc, edst, SS, SD, MC, DEN, EX, E, N);
    k_edge_agg   <<<ab, 256, 0, stream>>>(esrc, edst, FEAT, EX, DEN, AGG, E, N);
    k_finalize<<<nfBlocks, 256, 0, stream>>>(AGG, bias, g, bb, out_bf, out_f32, nodeFeat);
  };

  // ---- layer 1: in=128, out=4x64 ----
  const long long nx = (long long)N * 128;
  k_f32_to_bf16<<<(int)((nx + 255) / 256), 256, 0, stream>>>(x, BF, nx);
  k_f32_to_bf16<<<(256 * 128 + 255) / 256, 256, 0, stream>>>(W1, WB, 256 * 128);
  run_layer(128, asrc1, adst1, b1, bn1g, bn1b, BF, nullptr);   // bf16 x2 -> BF

  // ---- layer 2: in=256, out=4x64 ----
  k_f32_to_bf16<<<(256 * 256 + 255) / 256, 256, 0, stream>>>(W2, WB, 256 * 256);
  run_layer(256, asrc2, adst2, b2, bn2g, bn2b, nullptr, FEAT); // f32 x2 -> FEAT

  // ---- layer 3: heads=1, out=1 ----
  const int nb3 = (int)(((long long)N * 32 + 255) / 256);
  k_node3<<<nb3, 256, 0, stream>>>(FEAT, W3, asrc3, adst3, F3, SS3, SD3, MC3, DEN3, AGG3, N);
  k_edge_max1   <<<eb, 256, 0, stream>>>(esrc, edst, SS3, SD3, MC3, E, N);
  k_edge_expsum1<<<eb, 256, 0, stream>>>(esrc, edst, SS3, SD3, MC3, DEN3, EX3, E, N);
  k_edge_agg1   <<<eb, 256, 0, stream>>>(esrc, edst, F3, EX3, DEN3, AGG3, E, N);

  // ---- mean + classifier ----
  k_zero<<<1, 256, 0, stream>>>(RED, 256);
  int rb = (N + 255) / 256; if (rb > 1024) rb = 1024;
  k_reduce<<<rb, 256, 0, stream>>>(AGG3, RED, N);
  k_classifier<<<1, 64, 0, stream>>>(RED, b3, cW1, cb1, cW2, cb2, (float*)d_out, N);
}
